// unit_aagcn_68848325755539
// MI455X (gfx1250) — matrix operand order repaired and checked by reading, not compiled, not run on hardware
//
#include <hip/hip_runtime.h>
#include <hip/hip_bf16.h>

typedef __bf16 bhalf;
typedef __attribute__((ext_vector_type(16))) __bf16 v16bf;
typedef __attribute__((ext_vector_type(8)))  float  v8f;

#define N_  64
#define C_  64
#define T_  300
#define V_  25
#define S_  3
#define IC_ 16
#define OC_ 64
#define TTILES 19   // ceil(300/16)

// ---- workspace layout (bytes); total ~824 KB ----
#define WS_AA     0                          // float [N][S][32][32] padded A_adp
#define WS_AA_SZ  (N_*S_*32*32*4)
#define WS_WAB    (WS_AA + WS_AA_SZ)         // bf16 [S][2][IC][C] BN-folded conv_a/b
#define WS_WAB_SZ (S_*2*IC_*C_*2)
#define WS_BAB    (WS_WAB + WS_WAB_SZ)       // float [S][2][IC]
#define WS_BAB_SZ (S_*2*IC_*4)
#define WS_WD     (WS_BAB + WS_BAB_SZ)       // bf16 [S][OC][C] BN-folded conv_d
#define WS_WD_SZ  (S_*OC_*C_*2)
#define WS_BD     (WS_WD + WS_WD_SZ)         // float [OC]

// ---------- fragment helpers: everything loads as 2x ds_load_b128 ----------
union FragU { v16bf bf; uint4 q[2]; };
union Pk    { uint4 q; __bf16 h[8]; };

__device__ __forceinline__ v16bf ld_frag(const bhalf* p0, const bhalf* p1) {
  FragU f;
  f.q[0] = *(const uint4*)p0;   // elements e=0..7
  f.q[1] = *(const uint4*)p1;   // elements e=8..15
  return f.bf;
}
__device__ __forceinline__ v16bf zero_frag() {
  FragU f; f.q[0] = make_uint4(0,0,0,0); f.q[1] = make_uint4(0,0,0,0); return f.bf;
}
// A-operand 16x32: row-major A[m*ld + k]; k = kh + e (kh = 16*(lane/16))
__device__ __forceinline__ v16bf frag_a(const bhalf* A, int ld, int lane) {
  int m = lane & 15, kh = (lane >> 4) << 4;
  const bhalf* p = A + m * ld + kh;
  return ld_frag(p, p + 8);
}
// B-operand 32x16 from k-contiguous (transposed) storage Bt[n*ld + k]; k = kb + e
__device__ __forceinline__ v16bf frag_b(const bhalf* Bt, int ld, int lane) {
  int n = lane & 15, kb = (lane >> 4) << 4;
  const bhalf* p = Bt + n * ld + kb;
  return ld_frag(p, p + 8);
}
__device__ __forceinline__ v8f v8f_zero() {
  v8f z;
#pragma unroll
  for (int e = 0; e < 8; ++e) z[e] = 0.0f;
  return z;
}
__device__ __forceinline__ v8f wmma_bf16(v16bf a, v16bf b, v8f c) {
  return __builtin_amdgcn_wmma_f32_16x16x32_bf16(false, a, false, b, (short)0, c, false, false);
}

// ---------- K0: fold BN into conv weights ----------
__global__ void k0_prep(
    const float* caw, const float* cab, const float* cbw, const float* cbb,
    const float* ag, const float* abeta, const float* am, const float* av,
    const float* bg, const float* bbeta, const float* bm, const float* bv,
    const float* cdw, const float* cdb,
    const float* g, const float* beta, const float* mean, const float* var,
    char* ws)
{
  int tid = threadIdx.x;
  bhalf* Wab = (bhalf*)(ws + WS_WAB);
  float* Bab = (float*)(ws + WS_BAB);
  bhalf* Wd  = (bhalf*)(ws + WS_WD);
  float* Bd  = (float*)(ws + WS_BD);
  for (int idx = tid; idx < S_*2*IC_*C_; idx += blockDim.x) {
    int c = idx & 63, i = (idx >> 6) & 15, abi = (idx >> 10) & 1, s = idx >> 11;
    const float* w  = abi ? cbw : caw;
    const float* gg = abi ? bg  : ag;
    const float* vv = abi ? bv  : av;
    float sc = gg[s*IC_+i] * rsqrtf(vv[s*IC_+i] + 1e-5f);
    Wab[idx] = (bhalf)(w[(s*IC_+i)*C_ + c] * sc);
  }
  for (int idx = tid; idx < S_*2*IC_; idx += blockDim.x) {
    int i = idx & 15, abi = (idx >> 4) & 1, s = idx >> 5;
    const float* bb = abi ? cbb : cab;
    const float* gg = abi ? bg  : ag;
    const float* vv = abi ? bv  : av;
    const float* mm = abi ? bm  : am;
    const float* be = abi ? bbeta : abeta;
    float sc = gg[s*IC_+i] * rsqrtf(vv[s*IC_+i] + 1e-5f);
    Bab[idx] = (bb[s*IC_+i] - mm[s*IC_+i]) * sc + be[s*IC_+i];
  }
  for (int idx = tid; idx < S_*OC_*C_; idx += blockDim.x) {
    int o = (idx >> 6) & 63;
    float gn = g[o] * rsqrtf(var[o] + 1e-5f);
    Wd[idx] = (bhalf)(cdw[idx] * gn);
  }
  for (int o = tid; o < OC_; o += blockDim.x) {
    float gn = g[o] * rsqrtf(var[o] + 1e-5f);
    float bs = cdb[o] + cdb[OC_ + o] + cdb[2*OC_ + o];
    Bd[o] = gn * (bs - mean[o]) + beta[o];
  }
}

// ---------- K1: fused a,b -> M -> A_adp; grid = N, 256 threads ----------
// LDS: Wab 12288 | xs [16t][25v][72c] 57600 | a2t/b2t 6x4096 | Msum 12288  = 106752 B
// Waves 0..5 compute: wave w owns s = w>>1 and t-pairs of parity w&1 (4 per chunk).
// Persistent regs per wave: 4 accumulator tiles (32 VGPRs) -> no spills.
__global__ void __launch_bounds__(256)
k1_adaptive(const float* __restrict__ x, const float* __restrict__ A,
            const float* __restrict__ alpha, char* ws)
{
  extern __shared__ char smem[];
  bhalf* Wab  = (bhalf*)smem;
  bhalf* xs   = (bhalf*)(smem + 12288);
  bhalf* ab2  = (bhalf*)(smem + 69888);
  float* Msum = (float*)(smem + 94464);
  int tid = threadIdx.x, lane = tid & 31, wid = tid >> 5;
  int n = blockIdx.x;
  const int hi8 = (lane >> 4) << 3;

  const bhalf* Wab_ws = (const bhalf*)(ws + WS_WAB);
  const float* Bab    = (const float*)(ws + WS_BAB);
  float*       Aa     = (float*)(ws + WS_AA);

  for (int i = tid; i < 6144; i += 256) Wab[i] = Wab_ws[i];
  for (int i = tid; i < 3072; i += 256) Msum[i] = 0.0f;
  bhalf* a2t = ab2 + wid * 2048;   // [32 v][32 k], k = tl*16 + i
  bhalf* b2t = a2t + 1024;         // [32 w][32 k]
  if (wid < 6)
    for (int i = lane; i < 2048; i += 32) a2t[i] = (bhalf)0.0f;  // pads stay zero

  int s = wid >> 1;
  v8f macc[4];
#pragma unroll
  for (int q = 0; q < 4; ++q) macc[q] = v8f_zero();

  for (int tt = 0; tt < TTILES; ++tt) {
    __syncthreads();
    int tbase = tt * 16;
    for (int idx = tid; idx < 64*400; idx += 256) {     // v fastest: coalesced
      int c = idx / 400, r = idx - c*400;
      int t = r / 25, v = r - t*25;
      int tg = tbase + t;
      float val = (tg < T_) ? x[((size_t)(n*64 + c)*T_ + tg)*V_ + v] : 0.0f;
      xs[(t*25 + v)*72 + c] = (bhalf)val;
    }
    __syncthreads();
    if (wid < 6) {
      for (int pq = 0; pq < 4; ++pq) {
        int p = (wid & 1) + 2*pq;
        if (tbase + 2*p + 1 < T_) {          // wave-uniform guard
          // ---- a,b: D[i, 2t x 25] = Wab'[s] (16x64) * x (64 x 50) ----
          for (int jt = 0; jt < 4; ++jt) {
            int col = jt*16 + (lane & 15);
            int kb  = (lane >> 4) << 4;
            bool ok = col < 50;
            int tl = 0, v = 0;
            v16bf bf0, bf1;
            if (ok) {
              tl = col / 25; v = col - tl*25;
              const bhalf* bp = &xs[((2*p + tl)*25 + v)*72];
              bf0 = ld_frag(bp + kb,      bp + kb + 8);
              bf1 = ld_frag(bp + 32 + kb, bp + 32 + kb + 8);
            } else { bf0 = zero_frag(); bf1 = zero_frag(); }
#pragma unroll
            for (int abi = 0; abi < 2; ++abi) {
              const bhalf* W = Wab + (s*2 + abi) * 1024;
              v8f acc = v8f_zero();
              acc = wmma_bf16(frag_a(W,      64, lane), bf0, acc);
              acc = wmma_bf16(frag_a(W + 32, 64, lane), bf1, acc);
              if (ok) {                       // packed b128 scatter of tanh(.)
                const float* bias = Bab + (s*2 + abi) * IC_;
                Pk pk;
#pragma unroll
                for (int r = 0; r < 8; ++r)
                  pk.h[r] = (bhalf)tanhf(acc[r] + bias[r + hi8]);
                bhalf* dst = abi ? b2t : a2t;
                *(uint4*)&dst[v*32 + tl*16 + hi8] = pk.q;
              }
            }
          }
          // ---- M[v,w] += a2^T b2 (K=32 = 2t x 16i), both operands b128 loads ----
#pragma unroll
          for (int vi = 0; vi < 2; ++vi)
#pragma unroll
            for (int wi = 0; wi < 2; ++wi)
              macc[vi*2 + wi] = wmma_bf16(frag_a(a2t + vi*16*32, 32, lane),
                                          frag_b(b2t + wi*16*32, 32, lane),
                                          macc[vi*2 + wi]);
        }
      }
    }
  }
  __syncthreads();
  if (wid < 6) {
#pragma unroll
    for (int vi = 0; vi < 2; ++vi)
#pragma unroll
      for (int wi = 0; wi < 2; ++wi)
#pragma unroll
        for (int r = 0; r < 8; ++r) {
          int v = vi*16 + r + hi8, w = wi*16 + (lane & 15);
          atomicAdd(&Msum[(s*32 + v)*32 + w], macc[vi*2 + wi][r]);
        }
  }
  __syncthreads();
  float al = alpha[0];
  for (int idx = tid; idx < 3072; idx += 256) {
    int w = idx & 31, v = (idx >> 5) & 31, ss = idx >> 10;
    float val = 0.0f;
    if (v < V_ && w < V_)
      val = A[(ss*V_ + v)*V_ + w] + tanhf(Msum[idx] * (1.0f/4800.0f)) * al;
    Aa[(size_t)n*3072 + idx] = val;
  }
}

// ---------- K2: z for all s -> LDS, then per-tile y = sum_s z*A_adp, fused epilogue ----
// grid=(19,64). LDS: Wd 24576 | Bd 256 | aalT 6144 | xs 57600 | zs3 196608 = 285184 B
// Phase 2 holds ONE v8f accumulator at a time -> no spills.
__global__ void __launch_bounds__(256)
k2_output(const float* __restrict__ x, const char* __restrict__ ws,
          float* __restrict__ out)
{
  extern __shared__ char smem[];
  bhalf* Wd   = (bhalf*)smem;
  float* Bd   = (float*)(smem + 24576);
  bhalf* aalT = (bhalf*)(smem + 24832);   // [3][32 w][32 v]  (B^T, k-contiguous)
  bhalf* xs   = (bhalf*)(smem + 30976);   // [16 t][25 v][72 c]
  bhalf* zs3  = (bhalf*)(smem + 88576);   // [3][64 o][16 t][32 v]
  int tid = threadIdx.x, lane = tid & 31, wid = tid >> 5;
  int tt = blockIdx.x, n = blockIdx.y;
  int tbase = tt * 16;
  const int hi8 = (lane >> 4) << 3;

  const bhalf* Wd_ws = (const bhalf*)(ws + WS_WD);
  const float* Bd_ws = (const float*)(ws + WS_BD);
  const float* Aa    = (const float*)(ws + WS_AA) + (size_t)n*3072;

  for (int i = tid; i < 12288; i += 256) Wd[i] = Wd_ws[i];
  for (int i = tid; i < 64;    i += 256) Bd[i] = Bd_ws[i];
  for (int i = tid; i < 3072;  i += 256) {
    int v = i & 31, w = (i >> 5) & 31, s = i >> 10;
    aalT[i] = (bhalf)Aa[s*1024 + v*32 + w];      // transpose on load
  }
  for (int i = tid; i < 49152; i += 256) ((unsigned int*)zs3)[i] = 0u;
  for (int idx = tid; idx < 64*400; idx += 256) {
    int c = idx / 400, r = idx - c*400;
    int t = r / 25, v = r - t*25;
    int tg = tbase + t;
    float val = (tg < T_) ? x[((size_t)(n*64 + c)*T_ + tg)*V_ + v] : 0.0f;
    xs[(t*25 + v)*72 + c] = (bhalf)val;
  }
  __syncthreads();

  // ---- phase 1: z[s][o][(t,v)] = Wd'[s] (64x64) * x (64 x 400); 300 tiles ----
  for (int q = wid; q < 300; q += 8) {             // wave-uniform loop
    int s = q / 100, r2 = q - s*100;
    int ot = r2 / 25, nt = r2 - ot*25;
    int col = nt*16 + (lane & 15);
    int tl = col / 25, v = col - tl*25;
    int kb = (lane >> 4) << 4;
    const bhalf* bp = &xs[(tl*25 + v)*72];
    v16bf bf0 = ld_frag(bp + kb,      bp + kb + 8);
    v16bf bf1 = ld_frag(bp + 32 + kb, bp + 32 + kb + 8);
    const bhalf* W = Wd + s*4096 + ot*16*64;
    v8f acc = v8f_zero();
    acc = wmma_bf16(frag_a(W,      64, lane), bf0, acc);
    acc = wmma_bf16(frag_a(W + 32, 64, lane), bf1, acc);
    bhalf* zp = zs3 + s*32768;
#pragma unroll
    for (int r = 0; r < 8; ++r) {
      int o = ot*16 + r + hi8;
      zp[(o*16 + tl)*32 + v] = (bhalf)acc[r];
    }
  }
  __syncthreads();

  // ---- phase 2: y[o,t,w] = sum_s z[s] * A_adp[s], fused epilogue ----
  for (int j = 0; j < 16; ++j) {
    int idx = wid*16 + j;
    int ot = idx >> 5, rem = idx & 31, tl = rem >> 1, wt = rem & 1;
    v8f acc = v8f_zero();
#pragma unroll
    for (int s = 0; s < S_; ++s) {
      v16bf af = frag_a(zs3 + s*32768 + (ot*256 + tl)*32, 512, lane);
      v16bf bf = frag_b(aalT + s*1024 + wt*16*32, 32, lane);
      acc = wmma_bf16(af, bf, acc);
    }
    int w  = wt*16 + (lane & 15);
    int tg = tbase + tl;
    if (w < V_ && tg < T_) {
      Pk rx;                                       // residual: 8 x bf16 in one b128
      rx.q = *(const uint4*)&xs[(tl*25 + w)*72 + ot*16 + hi8];
#pragma unroll
      for (int r = 0; r < 8; ++r) {
        int o = ot*16 + r + hi8;
        float val = acc[r] + Bd[o] + (float)rx.h[r];
        out[((size_t)(n*64 + o)*T_ + tg)*V_ + w] = fmaxf(val, 0.0f);
      }
    }
  }
}

extern "C" void kernel_launch(void* const* d_in, const int* in_sizes, int n_in,
                              void* d_out, int out_size, void* d_ws, size_t ws_size,
                              hipStream_t stream) {
  (void)in_sizes; (void)n_in; (void)out_size; (void)ws_size;
  const float* x     = (const float*)d_in[0];
  const float* A     = (const float*)d_in[1];
  const float* alpha = (const float*)d_in[2];
  const float* caw   = (const float*)d_in[3];
  const float* cab   = (const float*)d_in[4];
  const float* cbw   = (const float*)d_in[5];
  const float* cbb   = (const float*)d_in[6];
  const float* ag    = (const float*)d_in[7];
  const float* abeta = (const float*)d_in[8];
  const float* am    = (const float*)d_in[9];
  const float* av    = (const float*)d_in[10];
  const float* bg    = (const float*)d_in[11];
  const float* bbeta = (const float*)d_in[12];
  const float* bm    = (const float*)d_in[13];
  const float* bv    = (const float*)d_in[14];
  const float* cdw   = (const float*)d_in[15];
  const float* cdb   = (const float*)d_in[16];
  const float* g     = (const float*)d_in[17];
  const float* beta  = (const float*)d_in[18];
  const float* mean  = (const float*)d_in[19];
  const float* var   = (const float*)d_in[20];
  char* ws = (char*)d_ws;

  k0_prep<<<1, 256, 0, stream>>>(caw, cab, cbw, cbb, ag, abeta, am, av,
                                 bg, bbeta, bm, bv, cdw, cdb,
                                 g, beta, mean, var, ws);
  k1_adaptive<<<N_, 256, 106752, stream>>>(x, A, alpha, ws);
  k2_output<<<dim3(TTILES, N_), 256, 285184, stream>>>(x, ws, (float*)d_out);
}
